// ModalityPooling_8297876815926
// MI455X (gfx1250) — matrix operand order repaired and checked by reading, not compiled, not run on hardware
//
#include <hip/hip_runtime.h>
#include <hip/hip_bf16.h>
#include <math.h>

typedef __attribute__((ext_vector_type(16))) __bf16 v16bf;
typedef __attribute__((ext_vector_type(8)))  float  v8f;

#define NB   64      // number of graphs (B)
#define D    256     // feature dim
#define H    64      // attn hidden
#define SPL  16      // deterministic pooling splits

// ---------------------------------------------------------------------------
// Scorer: s[i] = relu(x[i] @ W1 + b1) @ w2 + b2   via v_wmma_f32_16x16x32_bf16
// One wave per 16-row tile; 8 waves / block; W1 kept transposed bf16 in LDS.
// ---------------------------------------------------------------------------
__global__ __launch_bounds__(256) void scorer_wmma(
    const float* __restrict__ x, const float* __restrict__ w1,
    const float* __restrict__ b1, const float* __restrict__ w2,
    const float* __restrict__ b2, float* __restrict__ s, int N)
{
    __shared__ __align__(64) __bf16 bw1[H * D];   // [n][k] transposed, 32 KB
    __shared__ float sw2[H];
    __shared__ float sb1[H];

    // cooperative preload of W1 (global row-major [k][H]) transposed -> bf16 LDS
    for (int idx = threadIdx.x; idx < H * D; idx += blockDim.x) {
        int n = idx / D, k = idx % D;
        bw1[idx] = (__bf16)w1[k * H + n];
    }
    if (threadIdx.x < H) {
        sw2[threadIdx.x] = w2[threadIdx.x];
        sb1[threadIdx.x] = b1[threadIdx.x];
    }
    __syncthreads();

    const float b2v  = b2[0];
    const int lane   = threadIdx.x & 31;
    const int wave   = threadIdx.x >> 5;
    const int mrow   = lane & 15;      // A-matrix row within tile
    const int hs     = lane >> 4;      // lane half-select
    const int tiles  = (N + 15) >> 4;

    for (int t = blockIdx.x * 8 + wave; t < tiles; t += gridDim.x * 8) {
        int row  = t * 16 + mrow;
        int rowc = row < N ? row : N - 1;
        const float* xr = x + (size_t)rowc * D;

        v8f acc[4];
        #pragma unroll
        for (int nt = 0; nt < 4; ++nt)
            #pragma unroll
            for (int j = 0; j < 8; ++j) acc[nt][j] = 0.0f;

        #pragma unroll
        for (int kt = 0; kt < 8; ++kt) {
            const int k0  = kt * 32;
            const int kb1 = k0 + (hs ? 16 : 0);       // K group 1 base
            const int kb2 = kb1 + 8;                  // K group 2 base
            const float4 f0 = *(const float4*)(xr + kb1);
            const float4 f1 = *(const float4*)(xr + kb1 + 4);
            const float4 f2 = *(const float4*)(xr + kb2);
            const float4 f3 = *(const float4*)(xr + kb2 + 4);
            v16bf a;
            a[0]  = (__bf16)f0.x; a[1]  = (__bf16)f0.y; a[2]  = (__bf16)f0.z; a[3]  = (__bf16)f0.w;
            a[4]  = (__bf16)f1.x; a[5]  = (__bf16)f1.y; a[6]  = (__bf16)f1.z; a[7]  = (__bf16)f1.w;
            a[8]  = (__bf16)f2.x; a[9]  = (__bf16)f2.y; a[10] = (__bf16)f2.z; a[11] = (__bf16)f2.w;
            a[12] = (__bf16)f3.x; a[13] = (__bf16)f3.y; a[14] = (__bf16)f3.z; a[15] = (__bf16)f3.w;

            #pragma unroll
            for (int nt = 0; nt < 4; ++nt) {
                // B fragment: 16 contiguous bf16 of transposed W1 row n
                const v16bf bfrag =
                    *(const v16bf*)(bw1 + (nt * 16 + mrow) * D + k0 + hs * 16);
                acc[nt] = __builtin_amdgcn_wmma_f32_16x16x32_bf16(
                    false, a, false, bfrag, (short)0, acc[nt], false, false);
            }
        }

        // epilogue: h = relu(acc + b1[n]); partial[r] = sum_n h*w2[n]
        float partial[8];
        #pragma unroll
        for (int r = 0; r < 8; ++r) partial[r] = 0.0f;
        #pragma unroll
        for (int nt = 0; nt < 4; ++nt) {
            const int n = nt * 16 + mrow;      // C-matrix column held by this lane
            const float w2n = sw2[n], b1n = sb1[n];
            #pragma unroll
            for (int r = 0; r < 8; ++r) {
                float h = acc[nt][r] + b1n;
                h = fmaxf(h, 0.0f);
                partial[r] += h * w2n;
            }
        }
        // reduce the 16 lanes within each half (masks stay inside the half)
        #pragma unroll
        for (int r = 0; r < 8; ++r) {
            #pragma unroll
            for (int mk = 1; mk < 16; mk <<= 1)
                partial[r] += __shfl_xor(partial[r], mk, 32);
        }
        // lanes 0..7 store rows 0..7, lanes 16..23 store rows 8..15
        const int r = lane & 15;
        if (r < 8) {
            int orow = t * 16 + hs * 8 + r;
            if (orow < N) s[orow] = partial[r] + b2v;
        }
    }
}

// ---------------------------------------------------------------------------
// Segment starts via binary search (batch is sorted)
// ---------------------------------------------------------------------------
__global__ void seg_starts(const int* __restrict__ batch, int N, int* __restrict__ starts)
{
    int b = threadIdx.x;
    if (b > NB) return;
    int lo = 0, hi = N;
    while (lo < hi) { int mid = (lo + hi) >> 1; if (batch[mid] < b) lo = mid + 1; else hi = mid; }
    starts[b] = lo;
}

// ---------------------------------------------------------------------------
// Per-segment softmax: s[i] <- exp(s[i]-max); den[b] = sum. Deterministic trees.
// ---------------------------------------------------------------------------
__global__ __launch_bounds__(256) void seg_softmax(
    const int* __restrict__ starts, float* __restrict__ s, float* __restrict__ den)
{
    __shared__ float red[256];
    const int b  = blockIdx.x;
    const int s0 = starts[b], s1 = starts[b + 1];
    float mx = -INFINITY;
    for (int i = s0 + threadIdx.x; i < s1; i += 256) mx = fmaxf(mx, s[i]);
    red[threadIdx.x] = mx; __syncthreads();
    for (int off = 128; off > 0; off >>= 1) {
        if (threadIdx.x < off) red[threadIdx.x] = fmaxf(red[threadIdx.x], red[threadIdx.x + off]);
        __syncthreads();
    }
    mx = red[0];
    if (s1 <= s0) mx = 0.0f;          // empty segment -> reference uses 0
    __syncthreads();
    float sum = 0.0f;
    for (int i = s0 + threadIdx.x; i < s1; i += 256) {
        float e = __expf(s[i] - mx);
        s[i] = e;
        sum += e;
    }
    red[threadIdx.x] = sum; __syncthreads();
    for (int off = 128; off > 0; off >>= 1) {
        if (threadIdx.x < off) red[threadIdx.x] += red[threadIdx.x + off];
        __syncthreads();
    }
    if (threadIdx.x == 0) den[b] = red[0];
}

// ---------------------------------------------------------------------------
// Pooling: deterministic split partial sums, then fixed-order finalize
// ---------------------------------------------------------------------------
__global__ __launch_bounds__(256) void pool_partial(
    const float* __restrict__ x, const float* __restrict__ e,
    const int* __restrict__ starts, float* __restrict__ part)   // [NB][SPL][D]
{
    const int b = blockIdx.x, sp = blockIdx.y, f = threadIdx.x;
    const int s0 = starts[b], s1 = starts[b + 1];
    const int len = s1 - s0;
    const int chunk = (len + SPL - 1) / SPL;
    int i0 = s0 + sp * chunk;
    int i1 = i0 + chunk; if (i1 > s1) i1 = s1;
    float acc = 0.0f;
    for (int i = i0; i < i1; ++i) acc += x[(size_t)i * D + f] * e[i];
    part[((size_t)b * SPL + sp) * D + f] = acc;
}

__global__ __launch_bounds__(256) void pool_final(
    const float* __restrict__ part, const float* __restrict__ den,
    float* __restrict__ pool)
{
    const int b = blockIdx.x, f = threadIdx.x;
    float acc = 0.0f;
    #pragma unroll
    for (int sp = 0; sp < SPL; ++sp) acc += part[((size_t)b * SPL + sp) * D + f];
    const float d = den[b];
    pool[b * D + f] = (d > 0.0f) ? acc / d : 0.0f;
}

// ---------------------------------------------------------------------------
// Heads (tiny 64x256 GEMMs)
// ---------------------------------------------------------------------------
__global__ __launch_bounds__(256) void mlp_head_k(
    const float* __restrict__ pool, const float* __restrict__ w1,
    const float* __restrict__ b1, const float* __restrict__ w2,
    const float* __restrict__ b2, float* __restrict__ out)
{
    __shared__ float row[D];
    __shared__ float hid[D];
    const int b = blockIdx.x, t = threadIdx.x;
    row[t] = pool[b * D + t];
    __syncthreads();
    float acc = b1[t];
    for (int k = 0; k < D; ++k) acc += row[k] * w1[k * D + t];
    hid[t] = fmaxf(acc, 0.0f);
    __syncthreads();
    acc = b2[t];
    for (int k = 0; k < D; ++k) acc += hid[k] * w2[k * D + t];
    out[b * D + t] = acc;
}

__global__ __launch_bounds__(256) void lin_head_k(
    const float* __restrict__ pool, const float* __restrict__ w,
    const float* __restrict__ bias, float* __restrict__ out)
{
    __shared__ float row[D];
    const int b = blockIdx.x, t = threadIdx.x;
    row[t] = pool[b * D + t];
    __syncthreads();
    float acc = bias[t];
    for (int k = 0; k < D; ++k) acc += row[k] * w[k * D + t];
    out[b * D + t] = acc;
}

// ---------------------------------------------------------------------------
extern "C" void kernel_launch(void* const* d_in, const int* in_sizes, int n_in,
                              void* d_out, int out_size, void* d_ws, size_t ws_size,
                              hipStream_t stream)
{
    const float* x_gene = (const float*)d_in[0];
    const float* x_cpg  = (const float*)d_in[1];
    const float* x_mir  = (const float*)d_in[2];
    const int*   bg     = (const int*)d_in[3];
    const int*   bc     = (const int*)d_in[4];
    const int*   bm     = (const int*)d_in[5];
    const int Ng = in_sizes[0] / D;
    const int Nc = in_sizes[1] / D;
    const int Nm = in_sizes[2] / D;

    const float* x_mod[3]   = { x_gene, x_cpg, x_mir };
    const int*   bat_mod[3] = { bg, bc, bm };
    const int    N_mod[3]   = { Ng, Nc, Nm };
    // params: pool_{gene,cpg,mir}: w1,b1,w2,b2 at indices 6..17
    // mrna: 18..21, cnv: 22..25, lin_cpg: 26,27, lin_mir: 28,29

    // ---- workspace carve-up ----
    float* ws = (float*)d_ws;
    size_t off = 0;
    float* s_mod[3];
    s_mod[0] = ws + off; off += (size_t)Ng;
    s_mod[1] = ws + off; off += (size_t)Nc;
    s_mod[2] = ws + off; off += (size_t)Nm;
    float* den  = ws + off; off += 3 * NB;
    float* part = ws + off; off += (size_t)NB * SPL * D;
    float* pools = ws + off; off += 3 * (size_t)NB * D;
    int* starts = (int*)(ws + off);                 // 3*(NB+1) ints

    float* pool_mod[3] = { pools, pools + NB * D, pools + 2 * NB * D };

    for (int mo = 0; mo < 3; ++mo) {
        const float* w1 = (const float*)d_in[6 + 4 * mo + 0];
        const float* b1 = (const float*)d_in[6 + 4 * mo + 1];
        const float* w2 = (const float*)d_in[6 + 4 * mo + 2];
        const float* b2 = (const float*)d_in[6 + 4 * mo + 3];
        const int N = N_mod[mo];
        int* st = starts + mo * (NB + 1);
        float* dn = den + mo * NB;

        const int tiles = (N + 15) >> 4;
        const int grid  = (tiles + 7) / 8;
        scorer_wmma<<<grid, 256, 0, stream>>>(x_mod[mo], w1, b1, w2, b2, s_mod[mo], N);
        seg_starts<<<1, 128, 0, stream>>>(bat_mod[mo], N, st);
        seg_softmax<<<NB, 256, 0, stream>>>(st, s_mod[mo], dn);
        pool_partial<<<dim3(NB, SPL), 256, 0, stream>>>(x_mod[mo], s_mod[mo], st, part);
        pool_final<<<NB, 256, 0, stream>>>(part, dn, pool_mod[mo]);
    }

    float* out = (float*)d_out;
    mlp_head_k<<<NB, 256, 0, stream>>>(pool_mod[0],
        (const float*)d_in[18], (const float*)d_in[19],
        (const float*)d_in[20], (const float*)d_in[21], out);
    mlp_head_k<<<NB, 256, 0, stream>>>(pool_mod[0],
        (const float*)d_in[22], (const float*)d_in[23],
        (const float*)d_in[24], (const float*)d_in[25], out + NB * D);
    lin_head_k<<<NB, 256, 0, stream>>>(pool_mod[1],
        (const float*)d_in[26], (const float*)d_in[27], out + 2 * NB * D);
    lin_head_k<<<NB, 256, 0, stream>>>(pool_mod[2],
        (const float*)d_in[28], (const float*)d_in[29], out + 3 * NB * D);
}
